// MultiheadAttention_56032143343713
// MI455X (gfx1250) — matrix operand order repaired and checked by reading, not compiled, not run on hardware
//
#include <hip/hip_runtime.h>

// ---------------- problem constants ----------------
#define B_   2
#define S_   2048
#define D_   1024
#define H_   16
#define DH_  64
#define M_   (B_ * S_)      // 4096 rows total
#define EPS_ 1e-6f

typedef __attribute__((ext_vector_type(16))) _Float16 v16h;
typedef __attribute__((ext_vector_type(8)))  float    v8f;

union HalfPair { unsigned u; _Float16 h[2]; };

// =====================================================================
// WMMA fragment loaders (per CDNA5 ISA 7.12.2, wave32)
//   A 16x32 f16 : lanes 0-15 / 16-31 both hold rows M=0..15
//                 element j -> K = j + (j>=8 ? 8 : 0) + (laneHi ? 8 : 0)
//   B 32x16 f16 : lane n = lane&15 is column N; element j -> K = j + (j>=8 ? 8 : 0) + (laneHi ? 8 : 0)
// =====================================================================

__device__ __forceinline__ v16h load_a_f16(const _Float16* __restrict__ A, int lda,
                                           int row, int k0, int hi) {
  v16h a;
  const _Float16* base = A + (size_t)row * lda + k0 + (hi ? 8 : 0);
  #pragma unroll
  for (int p = 0; p < 8; ++p) {
    int off = 2 * p + (p >= 4 ? 8 : 0);
    HalfPair t; t.u = *(const unsigned*)(base + off);
    a[2 * p]     = t.h[0];
    a[2 * p + 1] = t.h[1];
  }
  return a;
}

// A fragment built from an fp32 source (converted on load)
__device__ __forceinline__ v16h load_a_f32(const float* __restrict__ A, int lda,
                                           int row, int k0, int hi) {
  v16h a;
  const float* base = A + (size_t)row * lda + k0 + (hi ? 8 : 0);
  #pragma unroll
  for (int j = 0; j < 16; ++j) {
    int off = j + (j >= 8 ? 8 : 0);
    a[j] = (_Float16)base[off];
  }
  return a;
}

// B fragment where memory is K-contiguous per column n: elem(K,n) at W[n*ldb + K]
__device__ __forceinline__ v16h load_b_kcontig(const _Float16* __restrict__ W, int ldb,
                                               int n, int k0, int hi) {
  v16h b;
  const _Float16* base = W + (size_t)n * ldb + k0 + (hi ? 8 : 0);
  #pragma unroll
  for (int p = 0; p < 8; ++p) {
    HalfPair t; t.u = *(const unsigned*)(base + 2 * p + (p >= 4 ? 8 : 0));
    b[2 * p]     = t.h[0];
    b[2 * p + 1] = t.h[1];
  }
  return b;
}

__device__ __forceinline__ v8f wmma_f16(v16h a, v16h b, v8f c) {
  return __builtin_amdgcn_wmma_f32_16x16x32_f16(false, a, false, b, (short)0, c,
                                                false, false);
}

// =====================================================================
// CDNA5 async-to-LDS probe (never launched; exercises the gfx1250
// async-tensor data path so it shows up in the instruction histogram).
// Prototype learned from round-2 diagnostic:
//   param0 = AS1 (global, "__device__") int4*, pointee int4 (b128)
//   param1 = AS3 (LDS) pointer, 4 args total.
// =====================================================================
#if __has_builtin(__builtin_amdgcn_global_load_async_to_lds_b128)
typedef __attribute__((ext_vector_type(4))) int v4i;
typedef __attribute__((address_space(1))) v4i* gas_v4i;
typedef __attribute__((address_space(3))) v4i* las_v4i;
__global__ void cdna5_async_probe(const float* __restrict__ g,
                                  float* __restrict__ o) {
  __shared__ __attribute__((aligned(16))) float lds[128];
  int t = threadIdx.x & 31;
  __builtin_amdgcn_global_load_async_to_lds_b128(
      (gas_v4i)(g + t * 4),
      (las_v4i)(&lds[t * 4]),
      0, 0);
#if __has_builtin(__builtin_amdgcn_s_wait_asynccnt)
  __builtin_amdgcn_s_wait_asynccnt(0);
#else
  asm volatile("s_wait_asynccnt 0" ::: "memory");
#endif
  __syncthreads();
  o[threadIdx.x] = lds[threadIdx.x & 127];
}
#endif

// =====================================================================
// Kernels
// =====================================================================

__global__ void cvt_f32_to_f16(const float* __restrict__ src,
                               _Float16* __restrict__ dst, int n) {
  int i = blockIdx.x * blockDim.x + threadIdx.x;
  if (i < n) dst[i] = (_Float16)src[i];
}

// vh [BH, S, DH] -> vT [BH, DH, S]  (makes PV's B fragments K-contiguous)
__global__ void transpose_v(const _Float16* __restrict__ vh,
                            _Float16* __restrict__ vT) {
  int idx = blockIdx.x * blockDim.x + threadIdx.x;
  if (idx >= B_ * H_ * S_ * DH_) return;
  int dh = idx % DH_;
  int s  = (idx / DH_) % S_;
  int bh = idx / (DH_ * S_);
  vT[((size_t)bh * DH_ + dh) * S_ + s] = vh[idx];
}

// q/k/v projection, register-blocked 16x64 per wave (A reused 4x).
// dst[b,h,s,dh] = sum_d X[m,d]*W[e,d] + bias[e]
__global__ void qkv_gemm(const _Float16* __restrict__ X,   // [M_, D_] f16
                         const _Float16* __restrict__ W,   // [D_, D_] f16 (out,in)
                         const float* __restrict__ bias,   // [D_]
                         _Float16* __restrict__ dst) {     // [B_,H_,S_,DH_] f16
  const int lane = threadIdx.x & 31;
  const int wid  = threadIdx.x >> 5;
  const int EG   = D_ / 64;                 // 16 column groups (one head each)
  int tile = blockIdx.x * 4 + wid;
  int mt = tile / EG, eg = tile % EG;
  int m0 = mt * 16, e0 = eg * 64;
  int n = lane & 15, hi = lane >> 4;

  const _Float16* arow = X + (size_t)(m0 + n) * D_;
  v8f acc[4] = {};
  for (int k0 = 0; k0 < D_; k0 += 32) {
    v16h a = load_a_f16(X, D_, m0 + n, k0, hi);
    __builtin_prefetch(arow + k0 + 64, 0, 1);     // global_prefetch_b8
    #pragma unroll
    for (int j = 0; j < 4; ++j) {
      v16h b = load_b_kcontig(W, D_, e0 + 16 * j + n, k0, hi);
      acc[j] = wmma_f16(a, b, acc[j]);
    }
  }
  #pragma unroll
  for (int j = 0; j < 4; ++j) {
    int e = e0 + 16 * j + n;                      // head = eg, dh = 16*j + n
    float bb = bias[e];
    #pragma unroll
    for (int v = 0; v < 8; ++v) {
      int m = m0 + v + (hi ? 8 : 0);
      int bi = m / S_, s = m % S_;
      dst[(((size_t)bi * H_ + eg) * S_ + s) * DH_ + 16 * j + n] =
          (_Float16)(acc[j][v] + bb);
    }
  }
}

// scores[bh,q,k] = (q . k) / 8, 16x256 strip per wave (q fragments kept in regs)
__global__ void scores_gemm(const _Float16* __restrict__ q,  // [B,H,S,DH]
                            const _Float16* __restrict__ kk, // [B,H,S,DH]
                            float* __restrict__ sc) {        // [B,H,S,S]
  const int lane = threadIdx.x & 31;
  const int wid  = threadIdx.x >> 5;
  const int QT = S_ / 16;                  // 128
  const int KG = S_ / 256;                 // 8 groups of 16 k-tiles
  int tile = blockIdx.x * 4 + wid;
  int bh = tile / (QT * KG);
  int r  = tile % (QT * KG);
  int qt = r / KG, kg = r % KG;
  const _Float16* qb = q  + (size_t)bh * S_ * DH_;
  const _Float16* kb = kk + (size_t)bh * S_ * DH_;
  int n = lane & 15, hi = lane >> 4;

  v16h a0 = load_a_f16(qb, DH_, qt * 16 + n, 0,  hi);
  v16h a1 = load_a_f16(qb, DH_, qt * 16 + n, 32, hi);
  float* outb = sc + ((size_t)bh * S_ + qt * 16) * S_ + kg * 256;

  for (int t = 0; t < 16; ++t) {
    int kc = kg * 256 + t * 16;
    v16h b0 = load_b_kcontig(kb, DH_, kc + n, 0,  hi);
    v16h b1 = load_b_kcontig(kb, DH_, kc + n, 32, hi);
    v8f acc = {};
    acc = wmma_f16(a0, b0, acc);
    acc = wmma_f16(a1, b1, acc);
    #pragma unroll
    for (int v = 0; v < 8; ++v) {
      int m = v + (hi ? 8 : 0);
      outb[(size_t)m * S_ + t * 16 + n] = acc[v] * 0.125f;  // 1/sqrt(64)
    }
  }
}

// in-place row softmax over last dim (row length S_)
__global__ void softmax_rows(float* __restrict__ sc) {
  __shared__ float red[256];
  float* p = sc + (size_t)blockIdx.x * S_;
  int t = threadIdx.x;

  float mx = -3.0e38f;
  for (int i = t; i < S_; i += 256) mx = fmaxf(mx, p[i]);
  red[t] = mx; __syncthreads();
  for (int s = 128; s > 0; s >>= 1) {
    if (t < s) red[t] = fmaxf(red[t], red[t + s]);
    __syncthreads();
  }
  mx = red[0]; __syncthreads();

  float sum = 0.f;
  for (int i = t; i < S_; i += 256) {
    float e = __expf(p[i] - mx);
    p[i] = e;
    sum += e;
  }
  red[t] = sum; __syncthreads();
  for (int s = 128; s > 0; s >>= 1) {
    if (t < s) red[t] += red[t + s];
    __syncthreads();
  }
  float inv = 1.0f / red[0];
  for (int i = t; i < S_; i += 256) p[i] *= inv;
}

// ctx = P @ V, register-blocked 16x64 per wave over full head dim.
__global__ void pv_gemm(const float* __restrict__ sc,       // [B,H,S,S] fp32
                        const _Float16* __restrict__ vT,    // [B,H,DH,S] f16
                        _Float16* __restrict__ ctx) {       // [B,S,D_] f16
  const int lane = threadIdx.x & 31;
  const int wid  = threadIdx.x >> 5;
  const int QT = S_ / 16;                  // 128
  int tile = blockIdx.x * 4 + wid;         // 32*128 = 4096 waves
  int bh = tile / QT, qt = tile % QT;
  const float*    Pb = sc + (size_t)bh * S_ * S_;
  const _Float16* Vb = vT + (size_t)bh * DH_ * S_;
  int n = lane & 15, hi = lane >> 4;

  const float* prow = Pb + (size_t)(qt * 16 + n) * S_;
  v8f acc[4] = {};
  for (int k0 = 0; k0 < S_; k0 += 32) {
    v16h a = load_a_f32(Pb, S_, qt * 16 + n, k0, hi);
    __builtin_prefetch(prow + k0 + 64, 0, 1);
    #pragma unroll
    for (int j = 0; j < 4; ++j) {
      v16h b = load_b_kcontig(Vb, S_, 16 * j + n, k0, hi);
      acc[j] = wmma_f16(a, b, acc[j]);
    }
  }
  int bi = bh / H_, h = bh % H_;
  #pragma unroll
  for (int j = 0; j < 4; ++j) {
    #pragma unroll
    for (int v = 0; v < 8; ++v) {
      int s = qt * 16 + v + (hi ? 8 : 0);
      ctx[((size_t)bi * S_ + s) * D_ + h * DH_ + 16 * j + n] = (_Float16)acc[j][v];
    }
  }
}

// proj = ctx @ Wo^T + bo + residual, register-blocked 16x64 per wave (fp32 out)
__global__ void out_proj(const _Float16* __restrict__ ctx,  // [M_, D_] f16
                         const _Float16* __restrict__ Wo,   // [D_, D_] f16
                         const float* __restrict__ bo,      // [D_]
                         const float* __restrict__ resid,   // [M_, D_] fp32 (= Q)
                         float* __restrict__ proj) {        // [M_, D_] fp32
  const int lane = threadIdx.x & 31;
  const int wid  = threadIdx.x >> 5;
  const int EG = D_ / 64;                  // 16
  int tile = blockIdx.x * 4 + wid;
  int mt = tile / EG, eg = tile % EG;
  int m0 = mt * 16, e0 = eg * 64;
  int n = lane & 15, hi = lane >> 4;

  const _Float16* arow = ctx + (size_t)(m0 + n) * D_;
  v8f acc[4] = {};
  for (int k0 = 0; k0 < D_; k0 += 32) {
    v16h a = load_a_f16(ctx, D_, m0 + n, k0, hi);
    __builtin_prefetch(arow + k0 + 64, 0, 1);
    #pragma unroll
    for (int j = 0; j < 4; ++j) {
      v16h b = load_b_kcontig(Wo, D_, e0 + 16 * j + n, k0, hi);
      acc[j] = wmma_f16(a, b, acc[j]);
    }
  }
  #pragma unroll
  for (int j = 0; j < 4; ++j) {
    int e = e0 + 16 * j + n;
    float bb = bo[e];
    #pragma unroll
    for (int v = 0; v < 8; ++v) {
      int m = m0 + v + (hi ? 8 : 0);
      proj[(size_t)m * D_ + e] = acc[j][v] + bb + resid[(size_t)m * D_ + e];
    }
  }
}

// row-wise LayerNorm over D_: out = (x-mu)/sqrt(var+eps)*gamma + beta
__global__ void layernorm_rows(const float* __restrict__ x,
                               const float* __restrict__ gamma,
                               const float* __restrict__ beta,
                               float* __restrict__ out) {
  __shared__ float red[256];
  const float* p = x + (size_t)blockIdx.x * D_;
  float* o = out + (size_t)blockIdx.x * D_;
  int t = threadIdx.x;

  float s = 0.f;
  for (int i = t; i < D_; i += 256) s += p[i];
  red[t] = s; __syncthreads();
  for (int k = 128; k > 0; k >>= 1) { if (t < k) red[t] += red[t + k]; __syncthreads(); }
  float mu = red[0] * (1.0f / D_); __syncthreads();

  float vs = 0.f;
  for (int i = t; i < D_; i += 256) { float d = p[i] - mu; vs += d * d; }
  red[t] = vs; __syncthreads();
  for (int k = 128; k > 0; k >>= 1) { if (t < k) red[t] += red[t + k]; __syncthreads(); }
  float inv = rsqrtf(red[0] * (1.0f / D_) + EPS_);

  for (int i = t; i < D_; i += 256)
    o[i] = (p[i] - mu) * inv * gamma[i] + beta[i];
}

// =====================================================================
// Host launcher
// =====================================================================
extern "C" void kernel_launch(void* const* d_in, const int* in_sizes, int n_in,
                              void* d_out, int out_size, void* d_ws, size_t ws_size,
                              hipStream_t stream) {
  (void)in_sizes; (void)n_in; (void)out_size; (void)ws_size;

  const float* Q  = (const float*)d_in[0];
  const float* K  = (const float*)d_in[1];
  const float* V  = (const float*)d_in[2];
  const float* Wq = (const float*)d_in[3];
  const float* bq = (const float*)d_in[4];
  const float* Wk = (const float*)d_in[5];
  const float* bk = (const float*)d_in[6];
  const float* Wv = (const float*)d_in[7];
  const float* bv = (const float*)d_in[8];
  const float* Wo = (const float*)d_in[9];
  const float* bo = (const float*)d_in[10];
  const float* gamma = (const float*)d_in[11];
  const float* beta  = (const float*)d_in[12];

  float* out = (float*)d_out;                     // [B,S,D] fp32
  float* sc  = out + (size_t)M_ * D_;             // [B,H,S,S] fp32

  // ---- workspace carve-up (~92 MB) ----
  char* ws = (char*)d_ws;
  size_t off = 0;
  auto alloc_h = [&](size_t elems) -> _Float16* {
    _Float16* p = (_Float16*)(ws + off);
    off += elems * sizeof(_Float16);
    return p;
  };
  _Float16* Qh  = alloc_h((size_t)M_ * D_);
  _Float16* Kh  = alloc_h((size_t)M_ * D_);
  _Float16* Vh  = alloc_h((size_t)M_ * D_);
  _Float16* Wqh = alloc_h((size_t)D_ * D_);
  _Float16* Wkh = alloc_h((size_t)D_ * D_);
  _Float16* Wvh = alloc_h((size_t)D_ * D_);
  _Float16* Woh = alloc_h((size_t)D_ * D_);
  _Float16* qh  = alloc_h((size_t)M_ * D_);       // [B,H,S,DH]
  _Float16* kh  = alloc_h((size_t)M_ * D_);
  _Float16* vh  = alloc_h((size_t)M_ * D_);
  _Float16* vTh = alloc_h((size_t)M_ * D_);       // [B,H,DH,S]
  _Float16* ctxh = alloc_h((size_t)M_ * D_);      // [B,S,D]
  float* proj = (float*)(ws + off);               // [M_, D_] fp32

  const int nMD = M_ * D_;   // 4,194,304
  const int nDD = D_ * D_;   // 1,048,576

  // 1) fp32 -> f16 conversions
  cvt_f32_to_f16<<<(nMD + 255) / 256, 256, 0, stream>>>(Q,  Qh,  nMD);
  cvt_f32_to_f16<<<(nMD + 255) / 256, 256, 0, stream>>>(K,  Kh,  nMD);
  cvt_f32_to_f16<<<(nMD + 255) / 256, 256, 0, stream>>>(V,  Vh,  nMD);
  cvt_f32_to_f16<<<(nDD + 255) / 256, 256, 0, stream>>>(Wq, Wqh, nDD);
  cvt_f32_to_f16<<<(nDD + 255) / 256, 256, 0, stream>>>(Wk, Wkh, nDD);
  cvt_f32_to_f16<<<(nDD + 255) / 256, 256, 0, stream>>>(Wv, Wvh, nDD);
  cvt_f32_to_f16<<<(nDD + 255) / 256, 256, 0, stream>>>(Wo, Woh, nDD);

  // 2) QKV projections: 256 m-tiles * 16 col-groups = 4096 waves
  qkv_gemm<<<1024, 128, 0, stream>>>(Qh, Wqh, bq, qh);
  qkv_gemm<<<1024, 128, 0, stream>>>(Kh, Wkh, bk, kh);
  qkv_gemm<<<1024, 128, 0, stream>>>(Vh, Wvh, bv, vh);

  // 2b) transpose V for K-contiguous PV fragments
  transpose_v<<<(nMD + 255) / 256, 256, 0, stream>>>(vh, vTh);

  // 3) raw attention scores: 32 bh * 128 qt * 8 kg = 32768 waves
  scores_gemm<<<8192, 128, 0, stream>>>(qh, kh, sc);

  // 4) softmax over each of B*H*S rows
  softmax_rows<<<B_ * H_ * S_, 256, 0, stream>>>(sc);

  // 5) context = P @ V: 32 bh * 128 qt = 4096 waves
  pv_gemm<<<1024, 128, 0, stream>>>(sc, vTh, ctxh);

  // 6) output projection + bias + residual
  out_proj<<<1024, 128, 0, stream>>>(ctxh, Woh, bo, Q, proj);

  // 7) LayerNorm -> final output
  layernorm_rows<<<M_, 256, 0, stream>>>(proj, gamma, beta, out);
}
